// Decoder_51316269252610
// MI455X (gfx1250) — matrix operand order repaired and checked by reading, not compiled, not run on hardware
//
#include <hip/hip_runtime.h>
#include <hip/hip_bf16.h>
#include <math.h>

// ---------- types for WMMA ----------
typedef __attribute__((ext_vector_type(16))) _Float16 v16h;
typedef __attribute__((ext_vector_type(8)))  _Float16 half8;
typedef __attribute__((ext_vector_type(2)))  _Float16 half2v;
typedef __attribute__((ext_vector_type(8)))  float    v8f;

// ---------- problem constants ----------
#define NBOX 372
#define KGLY 373          // N_BOX + 1
#define NLOC 85
#define BATCH 2
#define IMGD 128

// decoder layer tables: DEC_CH = [64,128,64,32,16,8,3], spatial 1->2->4->8->16->32->64
constexpr int CI[6]   = {64, 128, 64, 32, 16, 8};
constexpr int CO[6]   = {128, 64, 32, 16, 8, 3};
constexpr int WIN[6]  = {1, 2, 4, 8, 16, 32};           // input spatial side
constexpr int MP[6]   = {512, 256, 128, 64, 32, 16};    // 4*co padded to 16
constexpr int KP[6]   = {64, 128, 64, 32, 32, 32};      // ci padded to 32 (weight rows)
constexpr int CIP[6]  = {64, 128, 64, 32, 16, 8};       // ci padded to 8 (LDS act stride)
constexpr int SPAD[6] = {16, 16, 16, 64, 256, 1024};    // spatial padded to 16 (LDS rows)
constexpr int WOFF[6] = {0, 32768, 65536, 73728, 75776, 76800}; // f16 offsets, total 77312
#define WTOT 77312

// ---------- ws layout (bytes) ----------
#define WS_WEIGHTS 0
#define WS_TRANS   (160*1024)
#define WS_FLAGS   (176*1024)
#define WS_SORT    (192*1024)
#define WS_COUNT   (208*1024)
#define WS_DECODED (256*1024)

__device__ __forceinline__ int loc_of(int k) {
  // recon_indices(): repeat loc indices per boxes-per-location
  if (k < 256) return k >> 2;                 // f=8, b=4
  if (k < 352) return 64 + (k - 256) / 6;     // f=4, b=6
  if (k < 368) return 80 + ((k - 352) >> 2);  // f=2, b=4
  if (k < 372) return 84;                     // f=1, b=4
  return 85;                                  // background slot
}

__device__ __forceinline__ v16h pack16(half8 lo, half8 hi) {
  v16h r;
#pragma unroll
  for (int i = 0; i < 8; ++i) { r[i] = lo[i]; r[i + 8] = hi[i]; }
  return r;
}

// ---------- K0: repack decoder weights into WMMA A-operand f16 layout ----------
// A_l[m][k] = W_l[k, o, p, q], m=(o*2+p)*2+q, row stride KP[l], zero padded.
__global__ void prep_weights_kernel(const float* w0, const float* w1, const float* w2,
                                    const float* w3, const float* w4, const float* w5,
                                    _Float16* A) {
  int gid = blockIdx.x * 256 + threadIdx.x;
  if (gid >= WTOT) return;
  int l = 5;
  if      (gid < WOFF[1]) l = 0;
  else if (gid < WOFF[2]) l = 1;
  else if (gid < WOFF[3]) l = 2;
  else if (gid < WOFF[4]) l = 3;
  else if (gid < WOFF[5]) l = 4;
  const float* w = (l == 0) ? w0 : (l == 1) ? w1 : (l == 2) ? w2 : (l == 3) ? w3 : (l == 4) ? w4 : w5;
  int rem = gid - WOFF[l];
  int kP = KP[l];
  int m = rem / kP, k = rem % kP;
  int ci = CI[l], co = CO[l];
  float val = 0.0f;
  if (m < 4 * co && k < ci) {
    int o = m >> 2, p = (m >> 1) & 1, q = m & 1;
    val = w[((k * co + o) * 2 + p) * 2 + q];   // w shape (ci, co, 2, 2) row-major
  }
  A[gid] = (_Float16)val;
}

// ---------- K1: present compaction + stable depth sort + STN transforms ----------
__global__ void setup_kernel(const float* z_where, const float* z_present, const float* z_depth,
                             int* flags, int* sorted, int* counts, float* trans) {
  int b = blockIdx.x;
  int t = threadIdx.x;
  __shared__ float sd[KGLY];
  __shared__ int   sp[KGLY];
  if (t < KGLY) {
    bool pres; float d;
    if (t < NBOX) {
      pres = (z_present[b * NBOX + t] == 1.0f);
      d = z_depth[b * NLOC + loc_of(t)];
    } else { pres = true; d = -1000.0f; }
    sp[t] = pres ? 1 : 0;
    sd[t] = pres ? d : -INFINITY;
    flags[b * KGLY + t] = sp[t];
    float xy0, xy1, wh0, wh1;
    if (t < NBOX) {
      const float* zw = z_where + (size_t)(b * NBOX + t) * 4;
      xy0 = zw[0]; xy1 = zw[1]; wh0 = zw[2]; wh1 = zw[3];
    } else { xy0 = 0.5f; xy1 = 0.5f; wh0 = 1.0f; wh1 = 1.0f; }
    float swx = 1.0f / wh0, swy = 1.0f / wh1;
    float sxx = (1.0f - 2.0f * xy0) * swx;
    float sxy = (1.0f - 2.0f * xy1) * swy;
    float* tr = trans + (size_t)(b * KGLY + t) * 4;
    tr[0] = swx; tr[1] = swy; tr[2] = sxx; tr[3] = sxy;
  }
  __syncthreads();
  if (t < KGLY && sp[t]) {
    float dk = sd[t];
    int r = 0;
    for (int j = 0; j < KGLY; ++j)
      if (sp[j]) { float dj = sd[j]; if (dj > dk || (dj == dk && j < t)) ++r; }
    sorted[b * KGLY + r] = t;   // stable descending-depth order (matches jnp.argsort(-depths))
  }
  if (t == 0) {
    int c = 0;
    for (int j = 0; j < KGLY; ++j) c += sp[j];
    counts[b] = c;
  }
}

// ---------- K2: per-glyph WMMA decoder (one workgroup = one glyph, 8 waves) ----------
__global__ __launch_bounds__(256) void decode_kernel(
    const float* z_what,
    const float* b0, const float* b1, const float* b2,
    const float* b3, const float* b4, const float* b5,
    const _Float16* Wf16, const int* flags, float* decoded) {
  int g = blockIdx.x;            // 0 .. B*K-1
  if (!flags[g]) return;         // skip absent glyphs (uniform branch)
  int b = g / KGLY, k = g % KGLY;

  __shared__ _Float16 bufA[8192];   // act buffers, f16, [spatial(padded16)][channel(ciP)]
  __shared__ _Float16 bufB[8192];
  __shared__ float biasS[128];

  int tid = threadIdx.x;
  int lane = tid & 31, wv = tid >> 5;
  int hf = lane >> 4, lm = lane & 15;

  // layer0 input: SPAD=16 rows x ciP=64, row 0 = z_what, rows 1..15 zero
  const float* z = z_what + (size_t)(b * 86 + loc_of(k)) * 64;
  for (int i = tid; i < 16 * 64; i += 256)
    bufA[i] = (i < 64) ? (_Float16)z[i] : (_Float16)0.0f;
  __syncthreads();

  float* dbase = decoded + (size_t)g * 12288;   // (3,64,64) f32
  const float* biasPtr[6] = {b0, b1, b2, b3, b4, b5};

#pragma unroll
  for (int l = 0; l < 6; ++l) {
    const int co = CO[l];
    const int wIn = WIN[l];
    const int S = wIn * wIn;
    const int Mp = MP[l], kP = KP[l], Kt = kP >> 5;
    const int ciP = CIP[l];
    const int coP = (l < 5) ? CIP[l + 1] : 0;
    _Float16* actIn  = (l & 1) ? bufB : bufA;
    _Float16* actOut = (l & 1) ? bufA : bufB;

    if (tid < co) biasS[tid] = biasPtr[l][tid];
    if (l < 5) {
      int tot = SPAD[l + 1] * coP;              // zero incl. spatial+channel padding
      for (int i = tid; i < tot; i += 256) actOut[i] = (_Float16)0.0f;
    }
    __syncthreads();

    const _Float16* A = Wf16 + WOFF[l];
    int Mt = Mp >> 4;
    int Nt = SPAD[l] >> 4;
    for (int t = wv; t < Mt * Nt; t += 8) {
      int mt = t % Mt, nt = t / Mt;
      v8f acc = {};
      int s = nt * 16 + lm;   // output-tile column = spatial index (padded rows are zero)
#pragma unroll
      for (int kt = 0; kt < Kt; ++kt) {
        // A tile: lane holds row m = mt*16+lm; chunks at K offsets 16*hf and 16*hf+8
        const _Float16* arow = A + (size_t)(mt * 16 + lm) * kP + kt * 32;
        half8 a0 = *(const half8*)(arow + 16 * hf);
        half8 a1 = *(const half8*)(arow + 16 * hf + 8);
        // B tile: lane holds column n = lm; K = 16*hf + j contiguous in act layout.
        // Unconditional DS loads from clamped in-range addresses; zero via cndmask
        // (condition constant-folds for layers where ciP >= 32).
        int c0 = kt * 32 + 16 * hf;
        bool v0 = (c0 < ciP);
        bool v1 = (c0 + 8 < ciP);
        half8 z8 = {};
        half8 t0 = *(const half8*)(actIn + s * ciP + (v0 ? c0 : 0));
        half8 t1 = *(const half8*)(actIn + s * ciP + (v1 ? c0 + 8 : 0));
        half8 bb0 = v0 ? t0 : z8;
        half8 bb1 = v1 ? t1 : z8;
        v16h av = pack16(a0, a1);
        v16h bv = pack16(bb0, bb1);
        acc = __builtin_amdgcn_wmma_f32_16x16x32_f16(
            false, av, false, bv, (short)0, acc, false, false);
      }
      // Epilogue: acc[j] and acc[j+4] (j = (p,q) position) hold consecutive output
      // channels o0, o0+1 at the same spatial position -> packed b32 stores.
      // For layers 0..4, M == Mp so every element is valid (no guard emitted).
      if (S >= 16 || s < S) {
        int hh = s / wIn, ww = s % wIn;
        int o0 = mt * 4 + 2 * hf;                 // even channel index
        float bl = biasS[o0];
        float bh = biasS[(o0 + 1 < co) ? (o0 + 1) : o0];
#pragma unroll
        for (int j = 0; j < 4; ++j) {
          int p = j >> 1, q = j & 1;
          int sn = (2 * hh + p) * (2 * wIn) + (2 * ww + q);
          float v0f = acc[j] + bl;
          float v1f = acc[j + 4] + bh;
          if (l == 5) {
            // co=3: hf==0 lanes own channels 0,1; hf==1 lanes own channel 2 only
            dbase[o0 * 4096 + sn] = 1.0f / (1.0f + __expf(-v0f));
            if (hf == 0) dbase[(o0 + 1) * 4096 + sn] = 1.0f / (1.0f + __expf(-v1f));
          } else {
            half2v pk;
            pk[0] = (_Float16)fmaxf(v0f, 0.0f);
            pk[1] = (_Float16)fmaxf(v1f, 0.0f);
            *(half2v*)(actOut + sn * coP + o0) = pk;
          }
        }
      }
    }
    __syncthreads();
  }
}

// ---------- K3: fused STN + depth-ordered composite (no recs materialization) ----------
__global__ __launch_bounds__(256) void composite_kernel(
    const int* sorted, const int* counts, const float* trans,
    const float* decoded, float* out) {
  // each block covers 256 pixels of a single batch image (16384 % 256 == 0)
  int b = (blockIdx.x << 8) >> 14;
  __shared__ float strn[KGLY * 4];
  __shared__ int   skid[KGLY];
  __shared__ int   snum;
  if (threadIdx.x == 0) snum = counts[b];
  __syncthreads();
  int n = snum;
  for (int i = threadIdx.x; i < n; i += 256) {
    int k = sorted[b * KGLY + i];
    skid[i] = k;
    const float* tr = trans + (size_t)(b * KGLY + k) * 4;
    strn[i * 4 + 0] = tr[0]; strn[i * 4 + 1] = tr[1];
    strn[i * 4 + 2] = tr[2]; strn[i * 4 + 3] = tr[3];
  }
  __syncthreads();

  int pid = blockIdx.x * 256 + threadIdx.x;
  int rem = pid & 16383;
  int py = rem >> 7, px = rem & 127;
  float bx = (2.0f * px + 1.0f) / 128.0f - 1.0f;
  float by = (2.0f * py + 1.0f) / 128.0f - 1.0f;
  float oc[3] = {0.0f, 0.0f, 0.0f};
  for (int t = 0; t < n; ++t) {
    if (oc[0] >= 0.001f && oc[1] >= 0.001f && oc[2] >= 0.001f) break;  // fully occluded
    float gx = strn[t * 4 + 0] * bx + strn[t * 4 + 2];
    float gy = strn[t * 4 + 1] * by + strn[t * 4 + 3];
    float ix = ((gx + 1.0f) * 64.0f - 1.0f) * 0.5f;
    float iy = ((gy + 1.0f) * 64.0f - 1.0f) * 0.5f;
    float ix0f = floorf(ix), iy0f = floorf(iy);
    int ix0 = (int)ix0f, iy0 = (int)iy0f;
    if (ix0 < -1 || ix0 > 63 || iy0 < -1 || iy0 > 63) continue;  // glyph doesn't cover pixel
    float fx = ix - ix0f, fy = iy - iy0f;
    float w00 = (1.0f - fy) * (1.0f - fx), w01 = (1.0f - fy) * fx;
    float w10 = fy * (1.0f - fx),          w11 = fy * fx;
    bool vx0 = (ix0 >= 0) & (ix0 < 64), vx1 = (ix0 + 1 >= 0) & (ix0 + 1 < 64);
    bool vy0 = (iy0 >= 0) & (iy0 < 64), vy1 = (iy0 + 1 >= 0) & (iy0 + 1 < 64);
    float m00 = (vy0 && vx0) ? w00 : 0.0f, m01 = (vy0 && vx1) ? w01 : 0.0f;
    float m10 = (vy1 && vx0) ? w10 : 0.0f, m11 = (vy1 && vx1) ? w11 : 0.0f;
    int cx0 = min(max(ix0, 0), 63), cx1 = min(max(ix0 + 1, 0), 63);
    int cy0 = min(max(iy0, 0), 63), cy1 = min(max(iy0 + 1, 0), 63);
    const float* img = decoded + (size_t)(b * KGLY + skid[t]) * 12288;
#pragma unroll
    for (int ch = 0; ch < 3; ++ch) {
      const float* pch = img + ch * 4096;
      float v = m00 * pch[cy0 * 64 + cx0] + m01 * pch[cy0 * 64 + cx1] +
                m10 * pch[cy1 * 64 + cx0] + m11 * pch[cy1 * 64 + cx1];
      oc[ch] += v * ((oc[ch] < 0.001f) ? 1.0f : 0.0f);
    }
  }
  size_t base = (size_t)b * 49152 + (size_t)py * 128 + px;
  out[base]           = fminf(fmaxf(oc[0], 0.0f), 1.0f);
  out[base + 16384]   = fminf(fmaxf(oc[1], 0.0f), 1.0f);
  out[base + 32768]   = fminf(fmaxf(oc[2], 0.0f), 1.0f);
}

extern "C" void kernel_launch(void* const* d_in, const int* in_sizes, int n_in,
                              void* d_out, int out_size, void* d_ws, size_t ws_size,
                              hipStream_t stream) {
  (void)out_size; (void)ws_size;
  const float* z_what    = (const float*)d_in[0];
  const float* z_where   = (const float*)d_in[1];
  const float* z_present = (const float*)d_in[2];
  const float* z_depth   = (const float*)d_in[3];

  const float* w[6]; const float* bb[6];
  if (n_in >= 16) {
    // pytree-flattened: w0,b0,w1,b1,...
    for (int i = 0; i < 6; ++i) { w[i] = (const float*)d_in[4 + 2 * i]; bb[i] = (const float*)d_in[5 + 2 * i]; }
  } else {
    // single packed buffer: w0,b0,w1,b1,... contiguous
    static const int wsz[6] = {64*128*4, 128*64*4, 64*32*4, 32*16*4, 16*8*4, 8*3*4};
    static const int bsz[6] = {128, 64, 32, 16, 8, 3};
    const float* p = (const float*)d_in[4];
    for (int i = 0; i < 6; ++i) { w[i] = p; p += wsz[i]; bb[i] = p; p += bsz[i]; }
  }
  (void)in_sizes;

  char* ws = (char*)d_ws;
  _Float16* Wf16  = (_Float16*)(ws + WS_WEIGHTS);
  float* trans    = (float*)(ws + WS_TRANS);
  int* flags      = (int*)(ws + WS_FLAGS);
  int* sorted     = (int*)(ws + WS_SORT);
  int* counts     = (int*)(ws + WS_COUNT);
  float* decoded  = (float*)(ws + WS_DECODED);
  float* outF     = (float*)d_out;

  prep_weights_kernel<<<(WTOT + 255) / 256, 256, 0, stream>>>(
      w[0], w[1], w[2], w[3], w[4], w[5], Wf16);
  setup_kernel<<<BATCH, 384, 0, stream>>>(
      z_where, z_present, z_depth, flags, sorted, counts, trans);
  decode_kernel<<<BATCH * KGLY, 256, 0, stream>>>(
      z_what, bb[0], bb[1], bb[2], bb[3], bb[4], bb[5], Wf16, flags, decoded);
  composite_kernel<<<(BATCH * IMGD * IMGD + 255) / 256, 256, 0, stream>>>(
      sorted, counts, trans, decoded, outF);
}
